// EvalNet_5927054868581
// MI455X (gfx1250) — matrix operand order repaired and checked by reading, not compiled, not run on hardware
//
#include <hip/hip_runtime.h>
#include <stdint.h>

#define N_NODES 16384
#define N_EDGES 262144
#define N_REL   9
#define D_IN    384
#define D_H     768
#define N_CLS   5

typedef __attribute__((ext_vector_type(16))) __bf16 v16bf;
typedef __attribute__((ext_vector_type(8)))  float  v8f;

__device__ __forceinline__ unsigned short f2bf(float f) {
  unsigned int u = __float_as_uint(f);
  u += 0x7FFFu + ((u >> 16) & 1u);          // round-to-nearest-even
  return (unsigned short)(u >> 16);
}
__device__ __forceinline__ float bf2f(unsigned short s) {
  return __uint_as_float(((unsigned int)s) << 16);
}

// ---------------------------------------------------------------------------
// f32 -> bf16 conversion kernels
// ---------------------------------------------------------------------------
__global__ void k_f32_to_bf16(const float* __restrict__ in,
                              unsigned short* __restrict__ out, size_t n) {
  size_t i = (size_t)blockIdx.x * blockDim.x + threadIdx.x;
  if (i < n) out[i] = f2bf(in[i]);
}
__global__ void k_relu_f32_to_bf16(const float* __restrict__ in,
                                   unsigned short* __restrict__ out, size_t n) {
  size_t i = (size_t)blockIdx.x * blockDim.x + threadIdx.x;
  if (i < n) out[i] = f2bf(fmaxf(in[i], 0.f));
}

// ---------------------------------------------------------------------------
// bf16 WMMA GEMM:  C[M,Nn] = A[M,K] * W[Nn,K]^T + bias   (A,W bf16; C f32/bf16)
// Per wave: 32(M) x 64(N) tile, K-step 32 -> 8x v_wmma_f32_16x16x32_bf16.
// Block = 8 waves (2 in M, 4 in N) -> 64 x 256 block tile.
//
// Fragment fills follow CDNA5 ISA 7.12.2 wave32 layouts:
//   A 16x32 bf16: lanes 0-15 row M=l hold K={k0+hi*8 .. +7} U {k0+16+hi*8 ..}
//   B 32x16 bf16: lane col N=l holds K={k0+hi*8 .. +7} U {k0+16+hi*8 ..}
// Both are exactly two 128-bit global loads per lane per fragment.
// ---------------------------------------------------------------------------
union FragAB { v16bf v; uint32_t u[8]; };

template <bool OUT_BF16, bool RELU>
__global__ __launch_bounds__(256) void k_gemm_bf16(
    const unsigned short* __restrict__ A,   // [M,K] bf16 row-major
    const unsigned short* __restrict__ W,   // [Nn,K] bf16 row-major (acts as B^T)
    const float* __restrict__ bias,         // [Nn]
    void* __restrict__ Cout,                // [M,Nn]
    int M, int K, int Nn) {
  const int lane = threadIdx.x & 31;
  const int wave = threadIdx.x >> 5;
  const int l  = lane & 15;
  const int hi = lane >> 4;
  const int wm = wave & 1;
  const int wn = wave >> 1;
  const int Mb = blockIdx.x * 64 + wm * 32;
  const int Nb = blockIdx.y * 256 + wn * 64;

  v8f acc[2][4];
  const v8f vz = {0.f, 0.f, 0.f, 0.f, 0.f, 0.f, 0.f, 0.f};
#pragma unroll
  for (int i = 0; i < 2; i++)
#pragma unroll
    for (int j = 0; j < 4; j++) acc[i][j] = vz;

  for (int k0 = 0; k0 < K; k0 += 32) {
    FragAB a[2], b[4];
    const int klo = k0 + hi * 8;
    const int khi = k0 + 16 + hi * 8;
#pragma unroll
    for (int i = 0; i < 2; i++) {
      const unsigned short* ap = A + (size_t)(Mb + i * 16 + l) * K;
      *reinterpret_cast<uint4*>(&a[i].u[0]) =
          *reinterpret_cast<const uint4*>(ap + klo);
      *reinterpret_cast<uint4*>(&a[i].u[4]) =
          *reinterpret_cast<const uint4*>(ap + khi);
    }
    const int kb = k0 + hi * 8;
#pragma unroll
    for (int j = 0; j < 4; j++) {
      const unsigned short* wp = W + (size_t)(Nb + j * 16 + l) * K + kb;
      *reinterpret_cast<uint4*>(&b[j].u[0]) =
          *reinterpret_cast<const uint4*>(wp);
      *reinterpret_cast<uint4*>(&b[j].u[4]) =
          *reinterpret_cast<const uint4*>(wp + 16);
    }
#pragma unroll
    for (int i = 0; i < 2; i++)
#pragma unroll
      for (int j = 0; j < 4; j++)
        acc[i][j] = __builtin_amdgcn_wmma_f32_16x16x32_bf16(
            false, a[i].v, false, b[j].v, (short)0, acc[i][j], false, false);
  }

  // C layout (ISA 7.12.2): VGPR e -> row e (lanes 0-15) / row e+8 (lanes 16-31)
#pragma unroll
  for (int i = 0; i < 2; i++) {
#pragma unroll
    for (int j = 0; j < 4; j++) {
      const int col = Nb + j * 16 + l;
      const float bv = bias[col];
#pragma unroll
      for (int e = 0; e < 8; e++) {
        const int row = Mb + i * 16 + e + hi * 8;
        float v = acc[i][j][e] + bv;
        if (RELU) v = fmaxf(v, 0.f);
        if (OUT_BF16)
          ((unsigned short*)Cout)[(size_t)row * Nn + col] = f2bf(v);
        else
          ((float*)Cout)[(size_t)row * Nn + col] = v;
      }
    }
  }
}

// ---------------------------------------------------------------------------
// Per-edge normalized dot:  norm[e] = <x[dst], x[src]> / norm_const[etype]
// One wave32 per edge; lane covers 12 floats (3x float4) of each row.
// ---------------------------------------------------------------------------
__global__ __launch_bounds__(256) void k_edge_norm(
    const float* __restrict__ x, const int* __restrict__ src,
    const int* __restrict__ dst, const int* __restrict__ etype,
    const float* __restrict__ normc, float* __restrict__ norm) {
  const int gw = (int)((blockIdx.x * blockDim.x + threadIdx.x) >> 5);
  if (gw >= N_EDGES) return;
  const int lane = threadIdx.x & 31;
  const int s = src[gw], d = dst[gw];
  const float4* ps = reinterpret_cast<const float4*>(x + (size_t)s * D_IN) + lane * 3;
  const float4* pd = reinterpret_cast<const float4*>(x + (size_t)d * D_IN) + lane * 3;
  float sum = 0.f;
#pragma unroll
  for (int t = 0; t < 3; t++) {
    float4 a = ps[t], b = pd[t];
    sum += a.x * b.x + a.y * b.y + a.z * b.z + a.w * b.w;
  }
#pragma unroll
  for (int off = 16; off > 0; off >>= 1) sum += __shfl_xor(sum, off, 32);
  if (lane == 0) norm[gw] = sum / normc[etype[gw]];
}

// ---------------------------------------------------------------------------
// Relation-filtered scatter:  aggr[dst] += norm[e] * h_r[src]   (h_r bf16)
// One wave per edge; lane covers 24 columns (3x 128-bit bf16 loads).
// ---------------------------------------------------------------------------
__global__ __launch_bounds__(256) void k_scatter_rel(
    int rel, const int* __restrict__ src, const int* __restrict__ dst,
    const int* __restrict__ etype, const float* __restrict__ norm,
    const unsigned short* __restrict__ h, float* __restrict__ aggr) {
  const int gw = (int)((blockIdx.x * blockDim.x + threadIdx.x) >> 5);
  if (gw >= N_EDGES) return;
  if (etype[gw] != rel) return;
  const int lane = threadIdx.x & 31;
  const float nv = norm[gw];
  const int s = src[gw], d = dst[gw];
  const unsigned short* hp = h + (size_t)s * D_H + lane * 24;
  float* ap = aggr + (size_t)d * D_H + lane * 24;
#pragma unroll
  for (int t = 0; t < 3; t++) {
    uint4 q = *reinterpret_cast<const uint4*>(hp + t * 8);
    const unsigned short* ss = reinterpret_cast<const unsigned short*>(&q);
#pragma unroll
    for (int e = 0; e < 8; e++) atomicAdd(&ap[t * 8 + e], nv * bf2f(ss[e]));
  }
}

// ---------------------------------------------------------------------------
// Plain scatter:  acc2[dst] += hm[src]    (f32)
// ---------------------------------------------------------------------------
__global__ __launch_bounds__(256) void k_scatter_add(
    const int* __restrict__ src, const int* __restrict__ dst,
    const float* __restrict__ hm, float* __restrict__ acc2) {
  const int gw = (int)((blockIdx.x * blockDim.x + threadIdx.x) >> 5);
  if (gw >= N_EDGES) return;
  const int lane = threadIdx.x & 31;
  const int s = src[gw], d = dst[gw];
  const float* hp = hm + (size_t)s * D_H + lane * 24;
  float* ap = acc2 + (size_t)d * D_H + lane * 24;
#pragma unroll
  for (int t = 0; t < 6; t++) {
    float4 q = *reinterpret_cast<const float4*>(hp + t * 4);
    atomicAdd(&ap[t * 4 + 0], q.x);
    atomicAdd(&ap[t * 4 + 1], q.y);
    atomicAdd(&ap[t * 4 + 2], q.z);
    atomicAdd(&ap[t * 4 + 3], q.w);
  }
}

// ---------------------------------------------------------------------------
// Pool: pooled[c] += sum over 64 rows of relu(acc2[row, c])
// ---------------------------------------------------------------------------
__global__ __launch_bounds__(256) void k_pool(const float* __restrict__ acc2,
                                              float* __restrict__ pooled) {
  const int t = threadIdx.x;
  const int row0 = blockIdx.x * 64;
  float p0 = 0.f, p1 = 0.f, p2 = 0.f;
  for (int rr = 0; rr < 64; rr++) {
    const float* rp = acc2 + (size_t)(row0 + rr) * D_H;
    p0 += fmaxf(rp[t], 0.f);
    p1 += fmaxf(rp[t + 256], 0.f);
    p2 += fmaxf(rp[t + 512], 0.f);
  }
  atomicAdd(&pooled[t], p0);
  atomicAdd(&pooled[t + 256], p1);
  atomicAdd(&pooled[t + 512], p2);
}

// ---------------------------------------------------------------------------
// Readout: out[c] = (pooled/N) . out_W[c] + out_b[c]
// ---------------------------------------------------------------------------
__global__ __launch_bounds__(256) void k_out(const float* __restrict__ pooled,
                                             const float* __restrict__ outW,
                                             const float* __restrict__ outB,
                                             float* __restrict__ out) {
  __shared__ float red[256];
  const int t = threadIdx.x;
  const float inv = 1.0f / (float)N_NODES;
  const float pv0 = pooled[t] * inv;
  const float pv1 = pooled[t + 256] * inv;
  const float pv2 = pooled[t + 512] * inv;
  for (int c = 0; c < N_CLS; c++) {
    const float* w = outW + (size_t)c * D_H;
    red[t] = pv0 * w[t] + pv1 * w[t + 256] + pv2 * w[t + 512];
    __syncthreads();
    for (int off = 128; off > 0; off >>= 1) {
      if (t < off) red[t] += red[t + off];
      __syncthreads();
    }
    if (t == 0) out[c] = red[0] + outB[c];
    __syncthreads();
  }
}

// ---------------------------------------------------------------------------
extern "C" void kernel_launch(void* const* d_in, const int* in_sizes, int n_in,
                              void* d_out, int out_size, void* d_ws,
                              size_t ws_size, hipStream_t stream) {
  (void)in_sizes; (void)n_in; (void)out_size; (void)ws_size;
  const float* x     = (const float*)d_in[0];
  const int*   ei    = (const int*)d_in[1];
  const int*   etype = (const int*)d_in[2];
  const float* relW  = (const float*)d_in[3];
  const float* relB  = (const float*)d_in[4];
  const float* normc = (const float*)d_in[5];
  const float* linW  = (const float*)d_in[6];
  const float* linB  = (const float*)d_in[7];
  const float* selfW = (const float*)d_in[8];
  const float* selfB = (const float*)d_in[9];
  const float* outW  = (const float*)d_in[10];
  const float* outB  = (const float*)d_in[11];
  const int* src = ei;             // edge_index[0] : message source
  const int* dst = ei + N_EDGES;   // edge_index[1] : aggregation target
  float* out = (float*)d_out;

  // workspace carve-up (~223 MB total), 256B aligned
  char* ws = (char*)d_ws;
  size_t off = 0;
  auto alloc = [&](size_t bytes) {
    size_t cur = off;
    off += (bytes + 255) & ~(size_t)255;
    return cur;
  };
  unsigned short* xb    = (unsigned short*)(ws + alloc((size_t)N_NODES * D_IN * 2));
  unsigned short* wrel  = (unsigned short*)(ws + alloc((size_t)N_REL * D_H * D_IN * 2));
  unsigned short* wlin  = (unsigned short*)(ws + alloc((size_t)D_H * D_H * 2));
  unsigned short* wself = (unsigned short*)(ws + alloc((size_t)D_H * D_H * 2));
  float*          norm  = (float*)(ws + alloc((size_t)N_EDGES * 4));
  unsigned short* hrel  = (unsigned short*)(ws + alloc((size_t)N_NODES * D_H * 2));
  float*          aggr  = (float*)(ws + alloc((size_t)N_NODES * D_H * 4));
  unsigned short* x1b   = (unsigned short*)(ws + alloc((size_t)N_NODES * D_H * 2));
  float*          hm    = (float*)(ws + alloc((size_t)N_NODES * D_H * 4));
  float*          acc2  = (float*)(ws + alloc((size_t)N_NODES * D_H * 4));
  float*          pooled = (float*)(ws + alloc((size_t)D_H * 4));

  hipMemsetAsync(aggr, 0, (size_t)N_NODES * D_H * 4, stream);
  hipMemsetAsync(pooled, 0, (size_t)D_H * 4, stream);

  // fp32 -> bf16 conversions (x and all GEMM weights)
  {
    size_t n = (size_t)N_NODES * D_IN;
    k_f32_to_bf16<<<(unsigned)((n + 255) / 256), 256, 0, stream>>>(x, xb, n);
    n = (size_t)N_REL * D_H * D_IN;
    k_f32_to_bf16<<<(unsigned)((n + 255) / 256), 256, 0, stream>>>(relW, wrel, n);
    n = (size_t)D_H * D_H;
    k_f32_to_bf16<<<(unsigned)((n + 255) / 256), 256, 0, stream>>>(linW, wlin, n);
    k_f32_to_bf16<<<(unsigned)((n + 255) / 256), 256, 0, stream>>>(selfW, wself, n);
  }

  const int edgeGrid = N_EDGES / 8;  // one wave32 per edge, 8 waves/block
  k_edge_norm<<<edgeGrid, 256, 0, stream>>>(x, src, dst, etype, normc, norm);

  // RelationAwareMP: per relation, WMMA GEMM h_r = x*W_r^T + b_r, then scatter
  dim3 g1(N_NODES / 64, D_H / 256);
  for (int r = 0; r < N_REL; r++) {
    k_gemm_bf16<true, false><<<g1, 256, 0, stream>>>(
        xb, wrel + (size_t)r * D_H * D_IN, relB + (size_t)r * D_H, hrel,
        N_NODES, D_IN, D_H);
    k_scatter_rel<<<edgeGrid, 256, 0, stream>>>(r, src, dst, etype, norm, hrel,
                                                aggr);
  }

  // x1 = relu(aggr) -> bf16 for second-stage GEMMs
  {
    size_t n = (size_t)N_NODES * D_H;
    k_relu_f32_to_bf16<<<(unsigned)((n + 255) / 256), 256, 0, stream>>>(aggr, x1b, n);
  }

  // self_x -> acc2 ; hm -> hm ; then acc2[dst] += hm[src]
  k_gemm_bf16<false, false><<<g1, 256, 0, stream>>>(x1b, wself, selfB, acc2,
                                                    N_NODES, D_H, D_H);
  k_gemm_bf16<false, false><<<g1, 256, 0, stream>>>(x1b, wlin, linB, hm,
                                                    N_NODES, D_H, D_H);
  k_scatter_add<<<edgeGrid, 256, 0, stream>>>(src, dst, hm, acc2);

  // pooled = sum_n relu(acc2[n]) ; out = (pooled/N) @ out_W^T + out_b
  k_pool<<<N_NODES / 64, 256, 0, stream>>>(acc2, pooled);
  k_out<<<1, 256, 0, stream>>>(pooled, outW, outB, out);
}
